// RotarySelfAttention_22179211117424
// MI455X (gfx1250) — compile-verified, matrix operand order checked by reading, not run on hardware
//
#include <hip/hip_runtime.h>
#include <hip/hip_bf16.h>

typedef __attribute__((ext_vector_type(16))) _Float16 v16h;
typedef __attribute__((ext_vector_type(8)))  _Float16 v8h;
typedef __attribute__((ext_vector_type(8)))  float    v8f;
typedef __attribute__((ext_vector_type(4)))  unsigned v4u;
typedef __attribute__((ext_vector_type(8)))  int      v8i;
typedef __attribute__((ext_vector_type(4)))  int      v4i;

#define SEQ     2048
#define HID     2048
#define NHEADS  16
#define HD      128
#define BATCH   2
#define MTOT    (BATCH * SEQ)   // 4096

// ---------------------------------------------------------------------------
// Async global->LDS copy (GLOBAL_LOAD_ASYNC_TO_LDS_B128, ASYNCcnt).
// ---------------------------------------------------------------------------
__device__ __forceinline__ void async_load_b128(unsigned lds_byte_addr,
                                                const void* sbase,
                                                unsigned byte_off) {
    asm volatile("global_load_async_to_lds_b128 %0, %1, %2"
                 :: "v"(lds_byte_addr), "v"(byte_off), "s"(sbase)
                 : "memory");
}
__device__ __forceinline__ void wait_async0() {
    asm volatile("s_wait_asynccnt 0x0" ::: "memory");
}
__device__ __forceinline__ void wait_ds0() {
    asm volatile("s_wait_dscnt 0x0" ::: "memory");
}
__device__ __forceinline__ unsigned lds_addr_of(const void* p) {
    return (unsigned)(uintptr_t)p;   // shared aperture lives in the high 32 bits
}

// ---------------------------------------------------------------------------
// Tensor Data Mover: load a 2D tile (tileW x tileH elements, 2-byte elements)
// from a strided tensor in global memory into LDS. D# per CDNA5 ISA:
//  group0: [1:0]=count=1, [63:32]=lds_addr, [120:64]=global_addr, [127:126]=2
//  group1: [17:16]=data_size(1->2B), [79:48]=tensor_dim0, [111:80]=tensor_dim1,
//          [127:112]=tile_dim0, [143:128]=tile_dim1, [207:160]=dim0_stride
// Issued wave-uniform from one wave; completion tracked with TENSORcnt.
// ---------------------------------------------------------------------------
__device__ __forceinline__ void tdm_load_2d(unsigned lds_addr, const void* gptr,
                                            unsigned tileW, unsigned tileH,
                                            unsigned tensorW, unsigned tensorH,
                                            unsigned strideElems) {
    unsigned long long ga = (unsigned long long)(uintptr_t)gptr;
    v4u g0;
    g0.x = 1u;                                     // count=1 (valid), user mode
    g0.y = lds_addr;                               // LDS byte address
    g0.z = (unsigned)ga;                           // global_addr[31:0]
    g0.w = (unsigned)((ga >> 32) & 0x01FFFFFFu) | (2u << 30);  // addr[56:32], type=2
    v8i g1;
    g1[0] = (int)(1u << 16);                       // wg_mask=0, data_size=1 (2B)
    g1[1] = (int)((tensorW & 0xFFFFu) << 16);      // tensor_dim0[15:0]
    g1[2] = (int)(((tensorW >> 16) & 0xFFFFu) | ((tensorH & 0xFFFFu) << 16));
    g1[3] = (int)(((tensorH >> 16) & 0xFFFFu) | ((tileW & 0xFFFFu) << 16));
    g1[4] = (int)(tileH & 0xFFFFu);                // tile_dim1 (tile_dim2=0)
    g1[5] = (int)strideElems;                      // tensor_dim0_stride[31:0]
    g1[6] = 0;                                     // stride[47:32]=0, dim1_stride lo=0
    g1[7] = 0;
    v4i z4 = {0, 0, 0, 0};
#if __clang_major__ >= 23
    v8i z8 = {0, 0, 0, 0, 0, 0, 0, 0};
    __builtin_amdgcn_tensor_load_to_lds(g0, g1, z4, z4, z8, 0);
#else
    __builtin_amdgcn_tensor_load_to_lds(g0, g1, z4, z4, 0);
#endif
}

// ---------------------------------------------------------------------------
// A/B fragment loader for v_wmma_f32_16x16x32_f16.
// 16-bit A 16x32 layout: lanes 0-15 hold K {0..7, 16..23}; lanes 16-31 hold
// K {8..15, 24..31}; row (A) / column (B) index = lane & 15. Two b128 reads.
// ---------------------------------------------------------------------------
__device__ __forceinline__ v16h load_frag(const _Float16* base, int stride, int kOff) {
    int lane = threadIdx.x & 31;
    const _Float16* p = base + (size_t)(lane & 15) * stride + kOff + ((lane >> 4) << 3);
    union { v16h v; v8h h[2]; } u;
    u.h[0] = *(const v8h*)(p);
    u.h[1] = *(const v8h*)(p + 16);
    return u.v;
}

__device__ __forceinline__ v8f wmma_f16(v16h a, v16h b, v8f c) {
    return __builtin_amdgcn_wmma_f32_16x16x32_f16(false, a, false, b, (short)0, c, false, false);
}

// ---------------------------------------------------------------------------
// 16-lane butterfly reductions in pure VALU via DPP (rows of 16 lanes).
// ---------------------------------------------------------------------------
__device__ __forceinline__ float dpp_step(float x, const int ctrl_sel) {
    union { float f; int i; } s, t;
    s.f = x;
    switch (ctrl_sel) {
    case 0: t.i = __builtin_amdgcn_update_dpp(s.i, s.i, 0xB1,  0xF, 0xF, false); break;
    case 1: t.i = __builtin_amdgcn_update_dpp(s.i, s.i, 0x4E,  0xF, 0xF, false); break;
    case 2: t.i = __builtin_amdgcn_update_dpp(s.i, s.i, 0x141, 0xF, 0xF, false); break;
    default: t.i = __builtin_amdgcn_update_dpp(s.i, s.i, 0x140, 0xF, 0xF, false); break;
    }
    return t.f;
}
__device__ __forceinline__ float red_max16(float v) {
    v = fmaxf(v, dpp_step(v, 0));
    v = fmaxf(v, dpp_step(v, 1));
    v = fmaxf(v, dpp_step(v, 2));
    v = fmaxf(v, dpp_step(v, 3));
    return v;
}
__device__ __forceinline__ float red_sum16(float v) {
    v += dpp_step(v, 0);
    v += dpp_step(v, 1);
    v += dpp_step(v, 2);
    v += dpp_step(v, 3);
    return v;
}

// ---------------------------------------------------------------------------
// fp32 -> f16 conversion (grid-stride)
// ---------------------------------------------------------------------------
__global__ void f32_to_f16_kernel(const float* __restrict__ in, _Float16* __restrict__ out, int n) {
    for (int i = blockIdx.x * blockDim.x + threadIdx.x; i < n; i += gridDim.x * blockDim.x)
        out[i] = (_Float16)in[i];
}

// ---------------------------------------------------------------------------
// WMMA GEMM: Y[m,n] = sum_k X[m,k] * W[n,k] + bias[n]
// Block tile 128(M) x 128(N), 256 threads = 8 waves, each wave 32x64.
// K staged 64 at a time by the Tensor Data Mover into double-buffered LDS;
// DMA for stage k+1 overlaps the 16 WMMAs/wave of stage k.
// MODE 0: fp32 output, row-major MxN. MODE 1: f16 scatter to (B,NH,S,HD).
// ---------------------------------------------------------------------------
#define GEMM_TILE_H (128 * 64)          // halves per LDS tile
template <int MODE>
__global__ __launch_bounds__(256) void gemm_wmma_kernel(
    const _Float16* __restrict__ X,   // M x K (row-major)
    const _Float16* __restrict__ W,   // N x K (row-major)
    const float*    __restrict__ bias,
    void*           __restrict__ out,
    int M, int N, int K) {
    __shared__ __align__(16) _Float16 As[2 * GEMM_TILE_H];
    __shared__ __align__(16) _Float16 Bs[2 * GEMM_TILE_H];

    const int lane  = threadIdx.x & 31;
    const int wave  = threadIdx.x >> 5;
    const int waveM = wave >> 1;      // 0..3 -> 32-row slice
    const int waveN = wave & 1;       // 0..1 -> 64-col slice
    const int m0 = blockIdx.y * 128;
    const int n0 = blockIdx.x * 128;

    const unsigned asBase = lds_addr_of(As);
    const unsigned bsBase = lds_addr_of(Bs);

    v8f acc[2][4] = {};

    // Prologue: DMA stage 0 into buffer 0 (single wave issues; TENSORcnt).
    if (wave == 0) {
        tdm_load_2d(asBase, X + (size_t)m0 * K, 64, 128, (unsigned)K, (unsigned)M, (unsigned)K);
        tdm_load_2d(bsBase, W + (size_t)n0 * K, 64, 128, (unsigned)K, (unsigned)N, (unsigned)K);
        __builtin_amdgcn_s_wait_tensorcnt(0);
    }
    __syncthreads();

    for (int kb = 0; kb < K; kb += 64) {
        const int cur = (kb >> 6) & 1;
        // Kick off DMA for the next stage into the other buffer.
        if (kb + 64 < K && wave == 0) {
            unsigned bufOff = (unsigned)((cur ^ 1) * GEMM_TILE_H * 2);
            tdm_load_2d(asBase + bufOff, X + (size_t)m0 * K + kb + 64,
                        64, 128, (unsigned)K, (unsigned)M, (unsigned)K);
            tdm_load_2d(bsBase + bufOff, W + (size_t)n0 * K + kb + 64,
                        64, 128, (unsigned)K, (unsigned)N, (unsigned)K);
        }
        const _Float16* Ab = As + cur * GEMM_TILE_H;
        const _Float16* Bb = Bs + cur * GEMM_TILE_H;

#pragma unroll
        for (int kc = 0; kc < 2; ++kc) {
            v16h af0 = load_frag(Ab + (waveM * 32) * 64,      64, 32 * kc);
            v16h af1 = load_frag(Ab + (waveM * 32 + 16) * 64, 64, 32 * kc);
            v16h bf[4];
#pragma unroll
            for (int nt = 0; nt < 4; ++nt)
                bf[nt] = load_frag(Bb + (waveN * 64 + nt * 16) * 64, 64, 32 * kc);
#pragma unroll
            for (int nt = 0; nt < 4; ++nt) {
                acc[0][nt] = wmma_f16(af0, bf[nt], acc[0][nt]);
                acc[1][nt] = wmma_f16(af1, bf[nt], acc[1][nt]);
            }
        }
        // Wait for the in-flight DMA, then block-wide release of both buffers.
        if (wave == 0) __builtin_amdgcn_s_wait_tensorcnt(0);
        __syncthreads();
    }

    // Epilogue. C/D layout: VGPR j -> row j (lanes 0-15) or j+8 (lanes 16-31),
    // column = lane & 15.
    const int halfSel = lane >> 4;
    const int col     = lane & 15;
#pragma unroll
    for (int mt = 0; mt < 2; ++mt) {
#pragma unroll
        for (int nt = 0; nt < 4; ++nt) {
#pragma unroll
            for (int j = 0; j < 8; ++j) {
                int row = m0 + waveM * 32 + mt * 16 + j + 8 * halfSel;
                int cn  = n0 + waveN * 64 + nt * 16 + col;
                float v = acc[mt][nt][j] + bias[cn];
                if (MODE == 0) {
                    ((float*)out)[(size_t)row * N + cn] = v;
                } else {
                    int bb = row >> 11, s = row & (SEQ - 1);
                    int hh = cn >> 7,  d = cn & (HD - 1);
                    ((_Float16*)out)[(((size_t)bb * NHEADS + hh) * SEQ + s) * HD + d] = (_Float16)v;
                }
            }
        }
    }
}

// ---------------------------------------------------------------------------
// RoPE in-place on Q and K, layout (B, NHEADS, S, HD).
// ---------------------------------------------------------------------------
__global__ void rope_kernel(_Float16* __restrict__ q, _Float16* __restrict__ k) {
    int tid = blockIdx.x * blockDim.x + threadIdx.x;
    if (tid >= BATCH * NHEADS * SEQ * 64) return;
    int d  = tid & 63;
    int s  = (tid >> 6) & (SEQ - 1);
    int bh = tid >> 17;
    size_t base = ((size_t)bh * SEQ + s) * HD;
    float freq = (float)s * __expf(-9.210340371976184f * (float)d * (1.0f / 64.0f));
    float c  = __cosf(freq);
    float sn = __sinf(freq);

    float q0 = (float)q[base + d], q1 = (float)q[base + d + 64];
    q[base + d]      = (_Float16)(q0 * c - q1 * sn);
    q[base + d + 64] = (_Float16)(q1 * c + q0 * sn);

    float k0 = (float)k[base + d], k1 = (float)k[base + d + 64];
    k[base + d]      = (_Float16)(k0 * c - k1 * sn);
    k[base + d + 64] = (_Float16)(k1 * c + k0 * sn);
}

// ---------------------------------------------------------------------------
// Flash attention, 4 waves/block, each wave owns 16 query rows.
// K tile staged via async DMA; V tile transposed through VGPRs; softmax row
// reductions in DPP; P restaged through LDS with a wave-local dscnt wait.
// ---------------------------------------------------------------------------
__global__ __launch_bounds__(128) void attn_kernel(
    const _Float16* __restrict__ qh, const _Float16* __restrict__ kh,
    const _Float16* __restrict__ vh, _Float16* __restrict__ ctx) {
    __shared__ __align__(16) _Float16 Kl[32 * HD];     // 32 keys x 128 (row-major)
    __shared__ __align__(16) _Float16 Vt[HD * 32];     // transposed: 128 d x 32 keys
    __shared__ __align__(16) _Float16 Pl[4][16 * 32];  // per-wave P staging

    const int lane = threadIdx.x & 31;
    const int wave = threadIdx.x >> 5;
    const int qt = blockIdx.x, h = blockIdx.y, b = blockIdx.z;
    const int bh = b * NHEADS + h;

    const _Float16* qp = qh + ((size_t)bh * SEQ + qt * 64 + wave * 16) * HD;
    const _Float16* kp = kh + (size_t)bh * SEQ * HD;   // uniform per block
    const _Float16* vp = vh + (size_t)bh * SEQ * HD;
    const unsigned klBase = lds_addr_of(Kl);

    // Q fragments (16 rows x 128 d as 4 K-chunks of 32), kept in registers
    v16h qa[4];
#pragma unroll
    for (int c = 0; c < 4; ++c) qa[c] = load_frag(qp, HD, 32 * c);

    v8f o[8] = {};
    float m[8], l[8];
#pragma unroll
    for (int j = 0; j < 8; ++j) { m[j] = -1e30f; l[j] = 0.0f; }

    const float scale = 0.08838834764831845f;  // 128^-0.5

    for (int kb = 0; kb < SEQ; kb += 32) {
        __syncthreads();
        // K block: 32x128 contiguous halves -> 512 async b128 DMAs
#pragma unroll
        for (int it = 0; it < 4; ++it) {
            int i = threadIdx.x + (it << 7);          // 0..511
            async_load_b128(klBase + (unsigned)(i * 16),
                            kp, (unsigned)((kb * HD + i * 8) * 2));
        }
        // V block: load row-major, store transposed (d-major) for B fragments
        for (int i = threadIdx.x; i < 512; i += 128) {
            int kk = i >> 4;
            int d8 = (i & 15) << 3;
            v8h x = *(const v8h*)(vp + (size_t)(kb + kk) * HD + d8);
#pragma unroll
            for (int j = 0; j < 8; ++j) Vt[(d8 + j) * 32 + kk] = x[j];
        }
        wait_async0();
        __syncthreads();

        // Scores: 16 q-rows x 32 keys; batch 4 fragment loads per 4 WMMAs
        v8f s0 = {}, s1 = {};
#pragma unroll
        for (int cc = 0; cc < 4; cc += 2) {
            v16h kf[4];
            kf[0] = load_frag(Kl,           HD, 32 * cc);
            kf[1] = load_frag(Kl + 16 * HD, HD, 32 * cc);
            kf[2] = load_frag(Kl,           HD, 32 * (cc + 1));
            kf[3] = load_frag(Kl + 16 * HD, HD, 32 * (cc + 1));
            s0 = wmma_f16(qa[cc],     kf[0], s0);
            s1 = wmma_f16(qa[cc],     kf[1], s1);
            s0 = wmma_f16(qa[cc + 1], kf[2], s0);
            s1 = wmma_f16(qa[cc + 1], kf[3], s1);
        }

        // Online softmax update per row (row = j + 8*laneHalf); DPP reductions
#pragma unroll
        for (int j = 0; j < 8; ++j) {
            float a0 = s0[j] * scale, a1 = s1[j] * scale;
            float mx = red_max16(fmaxf(a0, a1));
            float mn   = fmaxf(m[j], mx);
            float corr = __expf(m[j] - mn);
            float p0 = __expf(a0 - mn), p1 = __expf(a1 - mn);
            float rs = red_sum16(p0 + p1);
            l[j] = l[j] * corr + rs;
            m[j] = mn;
#pragma unroll
            for (int t = 0; t < 8; ++t) o[t][j] *= corr;
            int row = j + ((lane >> 4) << 3);
            Pl[wave][row * 32 + (lane & 15)]      = (_Float16)p0;
            Pl[wave][row * 32 + 16 + (lane & 15)] = (_Float16)p1;
        }
        // P buffer is wave-private; LDS is in-order per wave -> dscnt wait only
        wait_ds0();

        // O += P (16x32) * V (32x128): batch 4 B-fragment loads per 4 WMMAs
        v16h pa = load_frag(Pl[wave], 32, 0);
#pragma unroll
        for (int hb = 0; hb < 2; ++hb) {
            v16h bv[4];
#pragma unroll
            for (int t = 0; t < 4; ++t)
                bv[t] = load_frag(Vt + (hb * 4 + t) * 16 * 32, 32, 0);
#pragma unroll
            for (int t = 0; t < 4; ++t)
                o[hb * 4 + t] = wmma_f16(pa, bv[t], o[hb * 4 + t]);
        }
    }

    // Normalize and write ctx (B, S, NHEADS, HD)
    const int halfSel = lane >> 4;
    const int col     = lane & 15;
#pragma unroll
    for (int j = 0; j < 8; ++j) {
        float inv = 1.0f / l[j];
        int s = qt * 64 + wave * 16 + j + 8 * halfSel;
        _Float16* op = ctx + (((size_t)b * SEQ + s) * NHEADS + h) * HD;
#pragma unroll
        for (int t = 0; t < 8; ++t) op[t * 16 + col] = (_Float16)(o[t][j] * inv);
    }
}

// ---------------------------------------------------------------------------
extern "C" void kernel_launch(void* const* d_in, const int* in_sizes, int n_in,
                              void* d_out, int out_size, void* d_ws, size_t ws_size,
                              hipStream_t stream) {
    const float* x  = (const float*)d_in[0];
    const float* Wq = (const float*)d_in[1];
    const float* bq = (const float*)d_in[2];
    const float* Wk = (const float*)d_in[3];
    const float* bk = (const float*)d_in[4];
    const float* Wv = (const float*)d_in[5];
    const float* bv = (const float*)d_in[6];
    const float* Wo = (const float*)d_in[7];
    const float* bo = (const float*)d_in[8];

    const int NX = MTOT * HID;     // 8,388,608
    const int NW = HID * HID;      // 4,194,304

    _Float16* Xh   = (_Float16*)d_ws;
    _Float16* Wqh  = Xh   + NX;
    _Float16* Wkh  = Wqh  + NW;
    _Float16* Wvh  = Wkh  + NW;
    _Float16* Woh  = Wvh  + NW;
    _Float16* qh   = Woh  + NW;     // (B, NHEADS, S, HD)
    _Float16* kh2  = qh   + NX;
    _Float16* vh   = kh2  + NX;
    _Float16* ctxh = vh   + NX;     // (B, S, H)

    f32_to_f16_kernel<<<4096, 256, 0, stream>>>(x,  Xh,  NX);
    f32_to_f16_kernel<<<2048, 256, 0, stream>>>(Wq, Wqh, NW);
    f32_to_f16_kernel<<<2048, 256, 0, stream>>>(Wk, Wkh, NW);
    f32_to_f16_kernel<<<2048, 256, 0, stream>>>(Wv, Wvh, NW);
    f32_to_f16_kernel<<<2048, 256, 0, stream>>>(Wo, Woh, NW);

    dim3 gGemm(HID / 128, MTOT / 128);  // (16, 32)
    gemm_wmma_kernel<1><<<gGemm, 256, 0, stream>>>(Xh, Wqh, bq, qh,  MTOT, HID, HID);
    gemm_wmma_kernel<1><<<gGemm, 256, 0, stream>>>(Xh, Wkh, bk, kh2, MTOT, HID, HID);
    gemm_wmma_kernel<1><<<gGemm, 256, 0, stream>>>(Xh, Wvh, bv, vh,  MTOT, HID, HID);

    rope_kernel<<<(BATCH * NHEADS * SEQ * 64) / 256, 256, 0, stream>>>(qh, kh2);

    attn_kernel<<<dim3(SEQ / 64, NHEADS, BATCH), 128, 0, stream>>>(qh, kh2, vh, ctxh);

    gemm_wmma_kernel<0><<<gGemm, 256, 0, stream>>>(ctxh, Woh, bo, d_out, MTOT, HID, HID);
}
